// MultiModalAttention_85641647882457
// MI455X (gfx1250) — compile-verified, matrix operand order checked by reading, not run on hardware
//
#include <hip/hip_runtime.h>

typedef unsigned short u16;
typedef unsigned int   u32;
typedef __attribute__((ext_vector_type(16))) __bf16 v16bf;
typedef __attribute__((ext_vector_type(8)))  float  v8f;

#define N_HEADS 8
#define T_DIM   2048
#define M_DIM   256
#define B_DIM   4
#define C_DIM   512

__device__ __forceinline__ u16 f2bf(float x) {
  u32 u = __float_as_uint(x);
  u32 r = u + 0x7FFFu + ((u >> 16) & 1u);   // round-to-nearest-even
  return (u16)(r >> 16);
}
__device__ __forceinline__ float bf2f(u16 h) {
  return __uint_as_float(((u32)h) << 16);
}
// scale two packed bf16 by s (s = power of two -> exact)
__device__ __forceinline__ u32 scale2bf(u32 pair, float s) {
  float lo = bf2f((u16)pair) * s;
  float hi = bf2f((u16)(pair >> 16)) * s;
  return (u32)f2bf(lo) | ((u32)f2bf(hi) << 16);
}

union FragU { u32 u[8]; v16bf v; };

// Load a 16x32 bf16 WMMA A/B fragment from LDS.
// Element [row0 + L%16][k0 + k] at lds[(row0+L%16)*stride + k0 + k], per the
// CDNA5 16-bit A-matrix layout (lanes 0-15: K pairs 0..7 / 16..23 in VGPR 0-7,
// lanes 16-31: +8). B fragments use the same loader on a transposed-staged tile.
__device__ __forceinline__ v16bf ld_frag(const u16* lds, int row0, int stride, int k0) {
  int lane = threadIdx.x & 31;
  int r  = row0 + (lane & 15);
  int kb = (lane >> 4) << 3;       // 0 or 8
  const u32* p = (const u32*)lds;  // stride and k are even -> dword aligned
  int base = r * stride + k0 + kb;
  FragU f;
#pragma unroll
  for (int i = 0; i < 4; ++i) {
    f.u[i]     = p[(base +      2 * i) >> 1];
    f.u[i + 4] = p[(base + 16 + 2 * i) >> 1];
  }
  return f.v;
}

__device__ __forceinline__ v16bf ones_frag() {
  FragU f;
#pragma unroll
  for (int i = 0; i < 8; ++i) f.u[i] = 0x3F803F80u;  // bf16 1.0 x2
  return f.v;
}

__device__ __forceinline__ v8f wmma_bf(v16bf a, v16bf b, v8f c) {
  return __builtin_amdgcn_wmma_f32_16x16x32_bf16(false, a, false, b, (short)0, c, false, false);
}

// ---------------------------------------------------------------- elementwise
__global__ void cvt_f32_bf16_kernel(const float* __restrict__ in, u16* __restrict__ out, int n) {
  int i = blockIdx.x * blockDim.x + threadIdx.x;
  if (i < n) out[i] = f2bf(in[i]);
}

__global__ void combine_kernel(const u16* __restrict__ sg, const u16* __restrict__ cg,
                               const u16* __restrict__ sv, const u16* __restrict__ cv,
                               u16* __restrict__ z, int n) {
  int i = blockIdx.x * blockDim.x + threadIdx.x;
  if (i < n) {
    float v = bf2f(sg[i]) * bf2f(cv[i]) + bf2f(cg[i]) * bf2f(sv[i]);
    z[i] = f2bf(v);
  }
}

// ---------------------------------------------------------------- GEMM
// out[Nrows,Nout] = A[Nrows,K](bf16) @ W[K,Nout](bf16) + bias, optional sigmoid.
// 128x128 tile, BK=32; 8 waves in 4x2 grid, each wave owns a 32x64 sub-tile.
// Double-buffered LDS: next k-tile prefetched to registers during compute.
template <bool OUT_BF16, bool SIGMOID>
__global__ __launch_bounds__(256) void gemm_bf16_kernel(
    const u16* __restrict__ A, const u16* __restrict__ W, const float* __restrict__ bias,
    void* __restrict__ outv, int Nrows, int K, int Nout) {
  __shared__ __align__(16) u16 lA[2][128 * 34];  // [m][k], pad 34 -> conflict free
  __shared__ __align__(16) u16 lB[2][128 * 34];  // [n][k] (W tile stored transposed)
  int tid = threadIdx.x;
  int w = tid >> 5, lane = tid & 31;
  int wrow = w >> 1, wcol = w & 1;
  int rb = blockIdx.y * 128, nb = blockIdx.x * 128;
  (void)Nrows;

  uint2 ad[4], wd[4];
  auto fetch = [&](int k0) {
#pragma unroll
    for (int i = 0; i < 4; ++i) {
      int c = tid + (i << 8);
      int row = c >> 3, col = (c & 7) << 2;          // A: 128 rows x 32 k
      ad[i] = *(const uint2*)(A + ((size_t)(rb + row) * K + k0 + col));
      int kr = c >> 5, nc = (c & 31) << 2;           // W: 32 k x 128 n
      wd[i] = *(const uint2*)(W + ((size_t)(k0 + kr) * Nout + nb + nc));
    }
  };
  auto stage = [&](int buf) {
#pragma unroll
    for (int i = 0; i < 4; ++i) {
      int c = tid + (i << 8);
      int row = c >> 3, col = (c & 7) << 2;
      u32* q = (u32*)&lA[buf][row * 34 + col];
      q[0] = ad[i].x; q[1] = ad[i].y;
      int kr = c >> 5, nc = (c & 31) << 2;
      lB[buf][(nc + 0) * 34 + kr] = (u16)wd[i].x;
      lB[buf][(nc + 1) * 34 + kr] = (u16)(wd[i].x >> 16);
      lB[buf][(nc + 2) * 34 + kr] = (u16)wd[i].y;
      lB[buf][(nc + 3) * 34 + kr] = (u16)(wd[i].y >> 16);
    }
  };

  v8f acc[2][4];
#pragma unroll
  for (int i = 0; i < 2; ++i)
#pragma unroll
    for (int j = 0; j < 4; ++j)
#pragma unroll
      for (int r = 0; r < 8; ++r) acc[i][j][r] = 0.f;

  fetch(0);
  stage(0);
  __syncthreads();

  int nk = K >> 5;
  for (int kt = 0; kt < nk; ++kt) {
    int cur = kt & 1;
    bool more = (kt + 1) < nk;
    if (more) fetch((kt + 1) << 5);                  // prefetch next tile

    v16bf af[2], bfr[4];
#pragma unroll
    for (int i = 0; i < 2; ++i) af[i] = ld_frag(lA[cur], wrow * 32 + i * 16, 34, 0);
#pragma unroll
    for (int j = 0; j < 4; ++j) bfr[j] = ld_frag(lB[cur], wcol * 64 + j * 16, 34, 0);
#pragma unroll
    for (int i = 0; i < 2; ++i)
#pragma unroll
      for (int j = 0; j < 4; ++j) acc[i][j] = wmma_bf(af[i], bfr[j], acc[i][j]);

    if (more) stage(cur ^ 1);
    __syncthreads();
  }

  int hl = lane >> 4, nlo = lane & 15;
#pragma unroll
  for (int i = 0; i < 2; ++i)
#pragma unroll
    for (int j = 0; j < 4; ++j) {
      int gcol = nb + wcol * 64 + j * 16 + nlo;
      float bb = bias[gcol];
#pragma unroll
      for (int r = 0; r < 8; ++r) {
        int grow = rb + wrow * 32 + i * 16 + r + hl * 8;
        float v = acc[i][j][r] + bb;
        if constexpr (SIGMOID) v = 1.f / (1.f + __expf(-v));
        if constexpr (OUT_BF16)
          ((u16*)outv)[(size_t)grow * Nout + gcol] = f2bf(v);
        else
          ((float*)outv)[(size_t)grow * Nout + gcol] = v;
      }
    }
}

// ---------------------------------------------------------------- bias kernel
// bias[b,h,t] = mean_m( softmax_m(s)[m,t] * s[m,t] ),  s = scale * qy . kx^T
// Scale folded into qy staging. One WG per (b,h, 128 t-cols); online softmax over M.
__global__ __launch_bounds__(256) void bias_kernel(
    const u16* __restrict__ qkvx, const u16* __restrict__ qkvy,
    float* __restrict__ biasout) {
  __shared__ __align__(16) u16 lqy[256 * 66];
  __shared__ __align__(16) u16 lkx[128 * 66];
  const int T = T_DIM, M = M_DIM;
  const float scale = 0.125f;
  int tid = threadIdx.x, w = tid >> 5, lane = tid & 31;
  int hl = lane >> 4, nlo = lane & 15;
  int tb = blockIdx.x * 128;
  int bh = blockIdx.y, b = bh >> 3, h = bh & 7;

#pragma unroll
  for (int i = 0; i < 16; ++i) {   // stage qy [256 x 64], pre-scaled
    int c = tid + (i << 8);
    int row = c >> 4, col = (c & 15) << 2;
    uint2 d = *(const uint2*)(qkvy + ((size_t)(b * M + row) * 1536 + h * 64 + col));
    u32* q = (u32*)&lqy[row * 66 + col];
    q[0] = scale2bf(d.x, scale); q[1] = scale2bf(d.y, scale);
  }
#pragma unroll
  for (int i = 0; i < 8; ++i) {    // stage kx [128 x 64]
    int c = tid + (i << 8);
    int row = c >> 4, col = (c & 15) << 2;
    uint2 d = *(const uint2*)(qkvx + ((size_t)(b * T + tb + row) * 1536 + 512 + h * 64 + col));
    u32* q = (u32*)&lkx[row * 66 + col];
    q[0] = d.x; q[1] = d.y;
  }
  __syncthreads();

  v16bf bk0 = ld_frag(lkx, w * 16, 66, 0);
  v16bf bk1 = ld_frag(lkx, w * 16, 66, 32);
  float runmax = -1e30f, se = 0.f, ses = 0.f;

  for (int mt = 0; mt < 16; ++mt) {
    v16bf aq0 = ld_frag(lqy, mt * 16, 66, 0);
    v16bf aq1 = ld_frag(lqy, mt * 16, 66, 32);
    v8f sf;
#pragma unroll
    for (int r = 0; r < 8; ++r) sf[r] = 0.f;
    sf = wmma_bf(aq0, bk0, sf);
    sf = wmma_bf(aq1, bk1, sf);
    float cm = sf[0];
#pragma unroll
    for (int r = 1; r < 8; ++r) cm = fmaxf(cm, sf[r]);
    cm = fmaxf(cm, __shfl_xor(cm, 16));          // combine the two 8-row halves
    float nm = fmaxf(runmax, cm);
    float alpha = __expf(runmax - nm);
    runmax = nm;
    se *= alpha; ses *= alpha;
    float pse = 0.f, pses = 0.f;
#pragma unroll
    for (int r = 0; r < 8; ++r) {
      float e = __expf(sf[r] - nm);
      pse += e; pses += e * sf[r];
    }
    pse  += __shfl_xor(pse, 16);
    pses += __shfl_xor(pses, 16);
    se += pse; ses += pses;
  }
  if (hl == 0)
    biasout[(size_t)bh * T + tb + w * 16 + nlo] = ses / (se * (float)M);
}

// ---------------------------------------------------------------- flash attention
// Q from qkvx (qx, pre-scaled by 1/sqrt(D)). K/V rows from kvsrc at koff/voff.
// One WG = 128 query rows (8 waves x 16); key blocks of 64, double-buffered.
// Row sums of P computed with WMMA against an all-ones B fragment.
__global__ __launch_bounds__(256) void flash_kernel(
    const u16* __restrict__ qsrc, const u16* __restrict__ kvsrc,
    const float* __restrict__ biasp, u16* __restrict__ outp,
    int Nk, int koff, int voff, int causal) {
  __shared__ __align__(16) u16 lk[2][64 * 66];     // K block [s][d]
  __shared__ __align__(16) u16 lvt[2][64 * 66];    // V block transposed [d][s]
  __shared__ __align__(16) u16 lp[8 * 16 * 66];    // per-wave Q then P staging
  const int T = T_DIM;
  const float scale = 0.125f;
  int tid = threadIdx.x, w = tid >> 5, lane = tid & 31;
  int hl = lane >> 4, nlo = lane & 15;
  int qb = blockIdx.x;
  int bh = blockIdx.y, b = bh >> 3, h = bh & 7;
  int tq0 = qb * 128;
  u16* lpw = lp + w * (16 * 66);

  uint2 kd[4], vd[4];
  auto fetch_kv = [&](int kb) {
#pragma unroll
    for (int i = 0; i < 4; ++i) {
      int c = tid + (i << 8);
      int row = c >> 4, col = (c & 15) << 2;
      size_t rbase = (size_t)(b * Nk + (kb << 6) + row) * 1536 + h * 64 + col;
      kd[i] = *(const uint2*)(kvsrc + rbase + koff);
      vd[i] = *(const uint2*)(kvsrc + rbase + voff);
    }
  };
  auto stage_kv = [&](int buf) {
#pragma unroll
    for (int i = 0; i < 4; ++i) {
      int c = tid + (i << 8);
      int row = c >> 4, col = (c & 15) << 2;
      u32* q = (u32*)&lk[buf][row * 66 + col];
      q[0] = kd[i].x; q[1] = kd[i].y;
      lvt[buf][(col + 0) * 66 + row] = (u16)vd[i].x;
      lvt[buf][(col + 1) * 66 + row] = (u16)(vd[i].x >> 16);
      lvt[buf][(col + 2) * 66 + row] = (u16)vd[i].y;
      lvt[buf][(col + 3) * 66 + row] = (u16)(vd[i].y >> 16);
    }
  };

#pragma unroll
  for (int i = 0; i < 8; ++i) {    // each wave stages its own 16x64 Q tile, pre-scaled
    int c = lane + (i << 5);
    int row = c >> 4, col = (c & 15) << 2;
    int t = tq0 + w * 16 + row;
    uint2 d = *(const uint2*)(qsrc + ((size_t)(b * T + t) * 1536 + h * 64 + col));
    u32* q = (u32*)&lpw[row * 66 + col];
    q[0] = scale2bf(d.x, scale); q[1] = scale2bf(d.y, scale);
  }
  fetch_kv(0);
  stage_kv(0);
  __syncthreads();
  v16bf aq0 = ld_frag(lpw, 0, 66, 0);
  v16bf aq1 = ld_frag(lpw, 0, 66, 32);
  v16bf onesf = ones_frag();

  v8f o[4];
  float rmax[8], rsum[8];
#pragma unroll
  for (int jj = 0; jj < 4; ++jj)
#pragma unroll
    for (int r = 0; r < 8; ++r) o[jj][r] = 0.f;
#pragma unroll
  for (int r = 0; r < 8; ++r) { rmax[r] = -1e30f; rsum[r] = 0.f; }

  int nblk = causal ? (qb * 2 + 2) : (Nk >> 6);

  for (int kb = 0; kb < nblk; ++kb) {
    int cur = kb & 1;
    bool more = (kb + 1) < nblk;
    if (more) fetch_kv(kb + 1);                      // prefetch next K/V block

    int s0 = kb << 6;
    float vals[4][8];
#pragma unroll
    for (int j = 0; j < 4; ++j) {  // S = Q K^T (16 x 64 per wave), pre-scaled Q
      v8f sf;
#pragma unroll
      for (int r = 0; r < 8; ++r) sf[r] = 0.f;
      v16bf bk0 = ld_frag(lk[cur], j * 16, 66, 0);
      v16bf bk1 = ld_frag(lk[cur], j * 16, 66, 32);
      sf = wmma_bf(aq0, bk0, sf);
      sf = wmma_bf(aq1, bk1, sf);
      int gs = s0 + j * 16 + nlo;
      float bv = biasp ? biasp[(size_t)bh * T + gs] : 0.f;
#pragma unroll
      for (int r = 0; r < 8; ++r) {
        float v = sf[r] + bv;
        if (causal) {
          int gt = tq0 + w * 16 + r + hl * 8;
          if (gs > gt) v = -1e30f;
        }
        vals[j][r] = v;
      }
    }
#pragma unroll
    for (int r = 0; r < 8; ++r) {  // online softmax: row max via shuffles only
      float bm = fmaxf(fmaxf(vals[0][r], vals[1][r]), fmaxf(vals[2][r], vals[3][r]));
#pragma unroll
      for (int m = 1; m < 16; m <<= 1) bm = fmaxf(bm, __shfl_xor(bm, m));
      float nm = fmaxf(rmax[r], bm);
      float alpha = __expf(rmax[r] - nm);
      rmax[r] = nm;
      rsum[r] *= alpha;
#pragma unroll
      for (int j = 0; j < 4; ++j) vals[j][r] = __expf(vals[j][r] - nm);
#pragma unroll
      for (int jj = 0; jj < 4; ++jj) o[jj][r] *= alpha;
    }
    // P (C-layout) -> per-wave LDS -> A-fragments for row-sum and P @ V
#pragma unroll
    for (int j = 0; j < 4; ++j)
#pragma unroll
      for (int r = 0; r < 8; ++r)
        lpw[(r + hl * 8) * 66 + j * 16 + nlo] = f2bf(vals[j][r]);
    asm volatile("s_wait_dscnt 0" ::: "memory");
    v16bf ap0 = ld_frag(lpw, 0, 66, 0);
    v16bf ap1 = ld_frag(lpw, 0, 66, 32);
    v8f psf;                       // row sums of P via WMMA against ones
#pragma unroll
    for (int r = 0; r < 8; ++r) psf[r] = 0.f;
    psf = wmma_bf(ap0, onesf, psf);
    psf = wmma_bf(ap1, onesf, psf);
#pragma unroll
    for (int r = 0; r < 8; ++r) rsum[r] += psf[r];
#pragma unroll
    for (int jj = 0; jj < 4; ++jj) {
      v16bf bv0 = ld_frag(lvt[cur], jj * 16, 66, 0);
      v16bf bv1 = ld_frag(lvt[cur], jj * 16, 66, 32);
      o[jj] = wmma_bf(ap0, bv0, o[jj]);
      o[jj] = wmma_bf(ap1, bv1, o[jj]);
    }

    if (more) stage_kv(cur ^ 1);
    __syncthreads();
  }

#pragma unroll
  for (int jj = 0; jj < 4; ++jj)
#pragma unroll
    for (int r = 0; r < 8; ++r) {
      int t = tq0 + w * 16 + r + hl * 8;
      int col = h * 64 + jj * 16 + nlo;
      outp[(size_t)(b * T + t) * 512 + col] = f2bf(o[jj][r] / rsum[r]);
    }
}

// ---------------------------------------------------------------- launch
extern "C" void kernel_launch(void* const* d_in, const int* in_sizes, int n_in,
                              void* d_out, int out_size, void* d_ws, size_t ws_size,
                              hipStream_t stream) {
  (void)in_sizes; (void)n_in; (void)out_size; (void)ws_size;
  const float* x     = (const float*)d_in[0];
  const float* y     = (const float*)d_in[1];
  // d_in[2] = attn_x_mask (causal tril) -> implemented analytically
  const float* Wqkvx = (const float*)d_in[3];
  const float* bqkvx = (const float*)d_in[4];
  const float* Wqkvy = (const float*)d_in[5];
  const float* bqkvy = (const float*)d_in[6];
  const float* Wgs   = (const float*)d_in[7];
  const float* bgs   = (const float*)d_in[8];
  const float* Wgc   = (const float*)d_in[9];
  const float* bgc   = (const float*)d_in[10];
  const float* Wp    = (const float*)d_in[11];
  const float* bp    = (const float*)d_in[12];

  char* p = (char*)d_ws;
  auto alloc = [&](size_t bytes) {
    char* r = p;
    p += (bytes + 255) & ~(size_t)255;
    return r;
  };
  const int BT = B_DIM * T_DIM;   // 8192
  const int BM = B_DIM * M_DIM;   // 1024
  u16*   xbf   = (u16*)alloc((size_t)BT * 512 * 2);
  u16*   ybf   = (u16*)alloc((size_t)BM * 512 * 2);
  u16*   wqx   = (u16*)alloc((size_t)512 * 1536 * 2);
  u16*   wqy   = (u16*)alloc((size_t)512 * 1536 * 2);
  u16*   wgs   = (u16*)alloc((size_t)512 * 512 * 2);
  u16*   wgc   = (u16*)alloc((size_t)512 * 512 * 2);
  u16*   wp    = (u16*)alloc((size_t)512 * 512 * 2);
  u16*   qkvx  = (u16*)alloc((size_t)BT * 1536 * 2);
  u16*   qkvy  = (u16*)alloc((size_t)BM * 1536 * 2);
  float* biasb = (float*)alloc((size_t)B_DIM * N_HEADS * T_DIM * 4);
  u16*   sval  = (u16*)alloc((size_t)BT * 512 * 2);
  u16*   cval  = (u16*)alloc((size_t)BT * 512 * 2);
  u16*   sgate = (u16*)alloc((size_t)BT * 512 * 2);
  u16*   cgate = (u16*)alloc((size_t)BT * 512 * 2);
  u16*   zbuf  = (u16*)alloc((size_t)BT * 512 * 2);

  auto cvt = [&](const float* in, u16* out, int n) {
    cvt_f32_bf16_kernel<<<(n + 255) / 256, 256, 0, stream>>>(in, out, n);
  };
  cvt(x, xbf, BT * 512);
  cvt(y, ybf, BM * 512);
  cvt(Wqkvx, wqx, 512 * 1536);
  cvt(Wqkvy, wqy, 512 * 1536);
  cvt(Wgs, wgs, 512 * 512);
  cvt(Wgc, wgc, 512 * 512);
  cvt(Wp,  wp,  512 * 512);

  // QKV projections (bf16 out)
  gemm_bf16_kernel<true, false><<<dim3(1536 / 128, BT / 128), 256, 0, stream>>>(
      xbf, wqx, bqkvx, qkvx, BT, 512, 1536);
  gemm_bf16_kernel<true, false><<<dim3(1536 / 128, BM / 128), 256, 0, stream>>>(
      ybf, wqy, bqkvy, qkvy, BM, 512, 1536);

  // cross-score bias over key positions
  bias_kernel<<<dim3(T_DIM / 128, B_DIM * N_HEADS), 256, 0, stream>>>(qkvx, qkvy, biasb);

  // self-attention (causal, +bias) and cross-attention
  flash_kernel<<<dim3(T_DIM / 128, B_DIM * N_HEADS), 256, 0, stream>>>(
      qkvx, qkvx, biasb, sval, T_DIM, 512, 1024, 1);
  flash_kernel<<<dim3(T_DIM / 128, B_DIM * N_HEADS), 256, 0, stream>>>(
      qkvx, qkvy, nullptr, cval, M_DIM, 512, 1024, 0);

  // gates (sigmoid fused, bf16 out)
  gemm_bf16_kernel<true, true><<<dim3(512 / 128, BT / 128), 256, 0, stream>>>(
      sval, wgs, bgs, sgate, BT, 512, 512);
  gemm_bf16_kernel<true, true><<<dim3(512 / 128, BT / 128), 256, 0, stream>>>(
      cval, wgc, bgc, cgate, BT, 512, 512);

  combine_kernel<<<(BT * 512 + 255) / 256, 256, 0, stream>>>(
      sgate, cgate, sval, cval, zbuf, BT * 512);

  // output projection (fp32 out to d_out)
  gemm_bf16_kernel<false, false><<<dim3(512 / 128, BT / 128), 256, 0, stream>>>(
      zbuf, wp, bp, d_out, BT, 512, 512);
}
